// VectorQuantizer_1443109012073
// MI455X (gfx1250) — matrix operand order repaired and checked by reading, not compiled, not run on hardware
//
#include <hip/hip_runtime.h>
#include <hip/hip_bf16.h>
#include <stdint.h>

// ---------------- problem constants ----------------
#define K_CODES   1024
#define D_DIM     64
#define N_VEC     131072            // 32*64*64
#define OUT_ELEMS (N_VEC * D_DIM)   // 8388608

// ---------------- kernel config --------------------
#define WG_THREADS   512            // 16 waves (wave32)
#define WAVES_PER_WG 16
#define ROWS_PER_WG  (WAVES_PER_WG * 16)   // 256 latent vectors per WG
#define CODE_CHUNK   256
#define N_CHUNKS     (K_CODES / CODE_CHUNK)
#define ROW_BYTES    144            // 64 bf16 (128B) + 16B pad -> 36-bank stride, 16B aligned

// LDS: emb chunk + enorm chunk + per-wave x tiles + per-wave inds scratch
#define LDS_EMB_OFF   0
#define LDS_ENORM_OFF (CODE_CHUNK * ROW_BYTES)                       // 36864
#define LDS_XTILE_OFF (LDS_ENORM_OFF + CODE_CHUNK * 4)               // 37888
#define LDS_INDS_OFF  (LDS_XTILE_OFF + WAVES_PER_WG * 16 * ROW_BYTES)// 74752
#define LDS_BYTES     (LDS_INDS_OFF + WAVES_PER_WG * 16 * 4)         // 75776

typedef __bf16 v16bf __attribute__((ext_vector_type(16)));
typedef float  v8f   __attribute__((ext_vector_type(8)));

union BF16Frag {
  v16bf v;
  uint4 q[2];
  unsigned u[8];
};

__device__ __forceinline__ unsigned pack_bf16_rne(float a, float b) {
  unsigned ua = __float_as_uint(a);
  unsigned ub = __float_as_uint(b);
  ua += 0x7fffu + ((ua >> 16) & 1u);   // round-to-nearest-even
  ub += 0x7fffu + ((ub >> 16) & 1u);
  return (ua >> 16) | (ub & 0xffff0000u);
}

__global__ void vq_init_ws(float* ws) {
  if (threadIdx.x == 0) ws[0] = 0.0f;
}

__global__ void vq_finish(const float* ws, float* lossSlot) {
  if (threadIdx.x == 0)
    lossSlot[0] = 0.75f * ws[0] / (float)OUT_ELEMS;   // (BETA+ALPHA)*mean((q-x)^2)
}

extern "C" __global__ __launch_bounds__(WG_THREADS)
void vq_main(const float* __restrict__ latents,   // [32,64,64,64] NCHW
             const float* __restrict__ embG,      // [1024,64]
             float* __restrict__ outF,            // [32,64,64,64] NCHW
             int*   __restrict__ indsOut,         // [131072]
             float* __restrict__ lossWs)
{
  extern __shared__ char lds[];
  char*  embL   = lds + LDS_EMB_OFF;
  float* enormL = (float*)(lds + LDS_ENORM_OFF);
  char*  xtileL = lds + LDS_XTILE_OFF;
  int*   indsL  = (int*)(lds + LDS_INDS_OFF);

  const int tid  = threadIdx.x;
  const int lane = tid & 31;
  const int wave = tid >> 5;
  const int hg   = lane >> 4;     // half-wave group: 0 or 1
  const int ln16 = lane & 15;

  // 16 consecutive flattened-NHWC rows for this wave: same (b,h), w = w0..w0+15
  const int rowBase = blockIdx.x * ROWS_PER_WG + wave * 16;
  const int bb  = rowBase >> 12;
  const int hh  = (rowBase >> 6) & 63;
  const int w0  = rowBase & 63;

  // ---- stage this wave's 16x64 x-tile (NCHW -> [w][c]) into LDS as bf16 ----
  char* xrowBase = xtileL + wave * 16 * ROW_BYTES;
  float xpart = 0.0f;
  {
    const float* src = latents + ((size_t)bb * 64) * 4096 + (size_t)hh * 64 + (w0 + ln16);
    char* dst = xrowBase + ln16 * ROW_BYTES;
    #pragma unroll
    for (int i = 0; i < 16; ++i) {
      const int c0 = i * 4 + hg * 2;                 // this lane covers channels {c0, c0+1}
      const float f0 = src[(size_t)c0 * 4096];
      const float f1 = src[(size_t)(c0 + 1) * 4096];
      xpart = fmaf(f0, f0, fmaf(f1, f1, xpart));     // fp32 ||x||^2 partial
      *(unsigned*)(dst + c0 * 2) = pack_bf16_rne(f0, f1);
    }
  }

  // ---- A fragments for v_wmma_f32_16x16x32_bf16 (16-bit A 16x32 layout) ----
  // lane holds row m=ln16; halves 0..7 = K in [8g,8g+8), halves 8..15 = K in [16+8g,16+8g+8)
  BF16Frag a0, a1;
  {
    const char* xr = xrowBase + ln16 * ROW_BYTES;
    a0.q[0] = *(const uint4*)(xr + 16 * hg);          // K  8g..8g+7
    a0.q[1] = *(const uint4*)(xr + 32 + 16 * hg);     // K 16+8g..
    a1.q[0] = *(const uint4*)(xr + 64 + 16 * hg);     // K 32+8g..
    a1.q[1] = *(const uint4*)(xr + 96 + 16 * hg);     // K 48+8g..
  }

  float bv[8];
  int   bi[8];
  #pragma unroll
  for (int i = 0; i < 8; ++i) { bv[i] = 3.4e38f; bi[i] = 0; }

  for (int chunk = 0; chunk < N_CHUNKS; ++chunk) {
    const int codeBase = chunk * CODE_CHUNK;

    __syncthreads();   // previous chunk's LDS reads done
    // ---- stage 256 codes: 8 steps x (32 rows x 16 threads/row), coalesced f32 reads ----
    #pragma unroll
    for (int s = 0; s < 8; ++s) {
      const int lrow = s * 32 + (tid >> 4);
      const int quad = tid & 15;
      const float4 f4 = *(const float4*)(embG + (size_t)(codeBase + lrow) * 64 + quad * 4);
      const unsigned p0 = pack_bf16_rne(f4.x, f4.y);
      const unsigned p1 = pack_bf16_rne(f4.z, f4.w);
      *(uint2*)(embL + lrow * ROW_BYTES + quad * 8) = make_uint2(p0, p1);
      float nr = fmaf(f4.x, f4.x, fmaf(f4.y, f4.y, fmaf(f4.z, f4.z, f4.w * f4.w)));
      nr += __shfl_xor(nr, 1, 32);
      nr += __shfl_xor(nr, 2, 32);
      nr += __shfl_xor(nr, 4, 32);
      nr += __shfl_xor(nr, 8, 32);
      if (quad == 0) enormL[lrow] = nr;              // fp32 ||e||^2
    }
    __syncthreads();

    // ---- 16 code-tiles of 16 codes each: two chained bf16 WMMAs (K=64) ----
    #pragma unroll 4
    for (int t = 0; t < 16; ++t) {
      const int lt = t * 16 + ln16;                  // local code row; B column n = ln16
      const char* er = embL + lt * ROW_BYTES;
      // 16-bit B 32x16 layout (matches A): halves 0..7 = K in [8g,8g+8), halves 8..15 = K in [16+8g,16+8g+8)
      BF16Frag fb0, fb1;
      fb0.q[0] = *(const uint4*)(er + 16 * hg);
      fb0.q[1] = *(const uint4*)(er + 32 + 16 * hg);
      fb1.q[0] = *(const uint4*)(er + 64 + 16 * hg);
      fb1.q[1] = *(const uint4*)(er + 96 + 16 * hg);

      v8f acc = {0.f, 0.f, 0.f, 0.f, 0.f, 0.f, 0.f, 0.f};
      acc = __builtin_amdgcn_wmma_f32_16x16x32_bf16(false, a0.v, false, fb0.v,
                                                    (short)0, acc, false, false);
      acc = __builtin_amdgcn_wmma_f32_16x16x32_bf16(false, a1.v, false, fb1.v,
                                                    (short)0, acc, false, false);

      const float en = enormL[lt];
      const int codeIdx = codeBase + lt;
      #pragma unroll
      for (int i = 0; i < 8; ++i) {
        const float s2 = fmaf(acc[i], -2.0f, en);    // ||e||^2 - 2 x.e
        if (s2 < bv[i]) { bv[i] = s2; bi[i] = codeIdx; }
      }
    }
  }

  // ---- cross-lane argmin over the 16-lane N groups (C layout: lanes0-15 rows0-7, 16-31 rows8-15)
  #pragma unroll
  for (int m = 1; m <= 8; m <<= 1) {
    #pragma unroll
    for (int i = 0; i < 8; ++i) {
      const float ov = __shfl_xor(bv[i], m, 32);
      const int   oi = __shfl_xor(bi[i], m, 32);
      if (ov < bv[i] || (ov == bv[i] && oi < bi[i])) { bv[i] = ov; bi[i] = oi; }
    }
  }

  // ---- indices + loss ( ||q-x||^2_row = ||x||^2_row + bestscore_row ) ----
  float lossLocal = xpart;
  if (ln16 == 0) {
    #pragma unroll
    for (int i = 0; i < 8; ++i) {
      const int row = hg * 8 + i;
      indsL[wave * 16 + row] = bi[i];
      indsOut[rowBase + row] = bi[i];
      lossLocal += bv[i];
    }
  }
  #pragma unroll
  for (int m = 1; m <= 16; m <<= 1) lossLocal += __shfl_xor(lossLocal, m, 32);
  if (lane == 0) atomicAdd(lossWs, lossLocal);

  // ---- gather fp32 codebook rows and write q in NCHW (w-coalesced per channel) ----
  const int myIdx = indsL[wave * 16 + ln16];         // same-wave LDS: in-order
  {
    float* outRow = outF + ((size_t)bb * 64) * 4096 + (size_t)hh * 64 + (w0 + ln16);
    const float* eRow = embG + (size_t)myIdx * 64;
    #pragma unroll
    for (int i = 0; i < 32; ++i) {
      const int c = i * 2 + hg;
      outRow[(size_t)c * 4096] = eRow[c];
    }
  }
}

extern "C" void kernel_launch(void* const* d_in, const int* in_sizes, int n_in,
                              void* d_out, int out_size, void* d_ws, size_t ws_size,
                              hipStream_t stream) {
  (void)in_sizes; (void)n_in; (void)out_size; (void)ws_size;
  const float* latents = (const float*)d_in[0];
  const float* emb     = (const float*)d_in[1];
  float* outF     = (float*)d_out;                 // [8388608] q in NCHW
  float* lossSlot = outF + OUT_ELEMS;              // [1] vq_loss
  int*   indsOut  = (int*)(outF + OUT_ELEMS + 1);  // [131072] argmin indices (int32)
  float* ws       = (float*)d_ws;

  vq_init_ws<<<1, 1, 0, stream>>>(ws);
  dim3 grid(N_VEC / ROWS_PER_WG);                  // 512 workgroups
  vq_main<<<grid, WG_THREADS, LDS_BYTES, stream>>>(latents, emb, outF, indsOut, ws);
  vq_finish<<<1, 1, 0, stream>>>(ws, lossSlot);
}
